// GraphLam_model_72447508349447
// MI455X (gfx1250) — matrix operand order repaired and checked by reading, not compiled, not run on hardware
//
#include <hip/hip_runtime.h>
#include <hip/hip_bf16.h>

// ---------------------------------------------------------------------------
// GraphLam (GraphCast-style GNN) for MI455X / gfx1250.
// Dense math: v_wmma_f32_16x16x32_f16 (f16 in, f32 accum), wave32.
// Weights live in LDS pre-swizzled into the hardware B-fragment layout
// (one v16h load -> 2x ds_load_b128). Layer-1 A fragments are built directly
// from global memory (gathers hit the 192MB L2); the hidden tile is scattered
// into the A-fragment layout so layer-2 A is a single v16h LDS load.
// ---------------------------------------------------------------------------

typedef __attribute__((ext_vector_type(16))) _Float16 v16h;
typedef __attribute__((ext_vector_type(8)))  float    v8f;

constexpr int WAVES = 2;            // waves per block
constexpr int BLOCK = WAVES * 32;
constexpr int HD    = 64;           // hidden dim
constexpr int TILES = 8;            // 16-row tiles per wave (amortizes weight staging)

__device__ inline float silu_f(float x) { return x / (1.f + __expf(-x)); }

// Load 8 consecutive features into fragment halves [h0..h0+7], bound-guarded
// (bound is a compile-time constant after inlining -> guards fold away).
__device__ inline void loadA8(const float* __restrict__ rowp, int c, int bound,
                              v16h& a, int h0) {
  if (c + 8 <= bound) {
    const float4 u = *(const float4*)(rowp + c);
    const float4 v = *(const float4*)(rowp + c + 4);
    a[h0+0]=(_Float16)u.x; a[h0+1]=(_Float16)u.y; a[h0+2]=(_Float16)u.z; a[h0+3]=(_Float16)u.w;
    a[h0+4]=(_Float16)v.x; a[h0+5]=(_Float16)v.y; a[h0+6]=(_Float16)v.z; a[h0+7]=(_Float16)v.w;
  } else {
#pragma unroll
    for (int j = 0; j < 8; ++j) {
      const int cc = c + j;
      a[h0+j] = (_Float16)((cc < bound) ? rowp[cc] : 0.f);
    }
  }
}

// Unguarded 8-float group (16B-aligned sources).
__device__ inline void loadA8f(const float* __restrict__ p, v16h& a, int h0) {
  const float4 u = *(const float4*)p;
  const float4 v = *(const float4*)(p + 4);
  a[h0+0]=(_Float16)u.x; a[h0+1]=(_Float16)u.y; a[h0+2]=(_Float16)u.z; a[h0+3]=(_Float16)u.w;
  a[h0+4]=(_Float16)v.x; a[h0+5]=(_Float16)v.y; a[h0+6]=(_Float16)v.z; a[h0+7]=(_Float16)v.w;
}

// ---- fused 2-layer MLP (+optional LayerNorm) on one 16-row tile per wave ----
// loadA(kc) returns the layer-1 A fragment for K-chunk kc.
// lWs1/lWs2: B-fragment-swizzled weight images; lHs: A-fragment-swizzled hidden.
template<int K1, bool DO_LN, int NOUT, typename F>
__device__ inline void mlp2_core(int lane, F&& loadA,
    const _Float16* __restrict__ lWs1, const float* __restrict__ lb1,
    const _Float16* __restrict__ lWs2, const float* __restrict__ lb2,
    const float* __restrict__ lg, const float* __restrict__ lbt,
    _Float16* __restrict__ lHs, float* __restrict__ lY, float* __restrict__ lR)
{
  constexpr int NP2 = (NOUT > 32) ? 64 : 32;
  constexpr int KC1 = K1 / 32, NT1 = 4, NT2 = NP2 / 16;
  const int c0 = lane & 15;
  const int r0 = (lane & 16) ? 8 : 0;

  // ---- layer 1: K1 -> 64, SiLU ----
  v8f acc[NT1] = {};
#pragma unroll
  for (int kc = 0; kc < KC1; ++kc) {
    const v16h a = loadA(kc);
#pragma unroll
    for (int nt = 0; nt < NT1; ++nt) {
      const v16h b = *((const v16h*)lWs1 + (kc * NT1 + nt) * 32 + lane);
      acc[nt] = __builtin_amdgcn_wmma_f32_16x16x32_f16(
          false, a, false, b, (short)0, acc[nt], false, false);
    }
  }
  // SiLU epilogue, scattered directly into the A-fragment layout of lHs.
#pragma unroll
  for (int nt = 0; nt < NT1; ++nt)
#pragma unroll
    for (int r = 0; r < 8; ++r) {
      const int row = r0 + r, col = nt * 16 + c0;
      const float v = silu_f(acc[nt][r] + lb1[col]);
      const int kc = col >> 5, ksub = col & 31, q = ksub >> 3;
      const int dl = row + ((q & 1) << 4);
      const int h  = ((q >> 1) << 3) + (ksub & 7);
      lHs[(kc * 32 + dl) * 16 + h] = (_Float16)v;  // wave-internal, no barrier
    }

  // ---- layer 2: 64 -> NOUT ----
  v8f acc2[NT2] = {};
#pragma unroll
  for (int kc = 0; kc < 2; ++kc) {
    const v16h a = *((const v16h*)lHs + kc * 32 + lane);
#pragma unroll
    for (int nt = 0; nt < NT2; ++nt) {
      const v16h b = *((const v16h*)lWs2 + (kc * NT2 + nt) * 32 + lane);
      acc2[nt] = __builtin_amdgcn_wmma_f32_16x16x32_f16(
          false, a, false, b, (short)0, acc2[nt], false, false);
    }
  }
#pragma unroll
  for (int nt = 0; nt < NT2; ++nt)
#pragma unroll
    for (int r = 0; r < 8; ++r) {
      const int row = r0 + r, col = nt * 16 + c0;
      lY[row * 64 + col] = acc2[nt][r] + lb2[col];
    }

  // ---- LayerNorm over 64 features, 2 lanes per row ----
  if (DO_LN) {
    const int row = lane & 15, hf = (lane >> 4) & 1;
    float s = 0.f, s2 = 0.f;
#pragma unroll
    for (int c = 0; c < 32; ++c) {
      const float v = lY[row * 64 + hf * 32 + c];
      s += v; s2 += v * v;
    }
    lR[hf * 16 + row]      = s;
    lR[32 + hf * 16 + row] = s2;
    const float ts  = lR[row] + lR[16 + row];
    const float ts2 = lR[32 + row] + lR[48 + row];
    const float mu  = ts * (1.f / 64.f);
    const float var = ts2 * (1.f / 64.f) - mu * mu;
    const float rs  = rsqrtf(var + 1e-5f);
#pragma unroll
    for (int c = 0; c < 32; ++c) {
      const int col = hf * 32 + c;
      const float v = lY[row * 64 + col];
      lY[row * 64 + col] = (v - mu) * rs * lg[col] + lbt[col];
    }
  }
}

// ---- weight staging: fp32 global -> f16 LDS, pre-swizzled to B fragments ---
template<int K1, int NOUT, bool HAS_LN>
__device__ inline void stage_weights(int tid,
    const float* __restrict__ W1, const float* __restrict__ b1,
    const float* __restrict__ W2, const float* __restrict__ b2,
    const float* __restrict__ g,  const float* __restrict__ bt,
    _Float16* lWs1, float* lb1, _Float16* lWs2, float* lb2, float* lg, float* lbt)
{
  constexpr int NP2 = (NOUT > 32) ? 64 : 32;
  constexpr int NT2 = NP2 / 16;
  for (int i = tid; i < K1 * HD; i += BLOCK) {
    const int k = i / HD, n = i % HD;
    const int kc = k >> 5, ksub = k & 31, nt = n >> 4;
    const int dl = (n & 15) + ((ksub & 8) ? 16 : 0), h = ((ksub & 16) >> 1) + (ksub & 7);
    lWs1[((kc * 4 + nt) * 32 + dl) * 16 + h] = (_Float16)W1[i];
  }
  for (int i = tid; i < HD * NP2; i += BLOCK) {
    const int k = i / NP2, n = i % NP2;
    const float v = (n < NOUT) ? W2[k * NOUT + n] : 0.f;
    const int kc = k >> 5, ksub = k & 31, nt = n >> 4;
    const int dl = (n & 15) + ((ksub & 8) ? 16 : 0), h = ((ksub & 16) >> 1) + (ksub & 7);
    lWs2[((kc * NT2 + nt) * 32 + dl) * 16 + h] = (_Float16)v;
  }
  for (int i = tid; i < HD; i += BLOCK) {
    lb1[i] = b1[i];
    lb2[i] = (i < NOUT) ? b2[i] : 0.f;
    if (HAS_LN) { lg[i] = g[i]; lbt[i] = bt[i]; }
  }
}

// ============================ kernels =======================================

// Node/edge embedder / encoding MLP: x[n][IN_RAW] -> out[n][64], LN.
template<int IN_RAW, int K1>
__global__ void __launch_bounds__(BLOCK) node_mlp_kernel(
    const float* __restrict__ x,
    const float* __restrict__ W1, const float* __restrict__ b1,
    const float* __restrict__ W2, const float* __restrict__ b2,
    const float* __restrict__ g,  const float* __restrict__ bt,
    float* __restrict__ out)
{
  __shared__ __align__(32) _Float16 lWs1[K1 * HD];
  __shared__ __align__(32) _Float16 lWs2[HD * HD];
  __shared__ __align__(32) _Float16 lHs[WAVES][2 * 32 * 16];
  __shared__ float lb1[HD], lb2[HD], lg[HD], lbt[HD];
  __shared__ float lY[WAVES][16 * 64];
  __shared__ float lR[WAVES][64];

  const int tid = threadIdx.x, lane = tid & 31, w = tid >> 5;
  stage_weights<K1, HD, true>(tid, W1, b1, W2, b2, g, bt, lWs1, lb1, lWs2, lb2, lg, lbt);
  __syncthreads();

  for (int t = 0; t < TILES; ++t) {
    const int base = ((blockIdx.x * WAVES + w) * TILES + t) * 16;
    const float* rowp = x + (size_t)(base + (lane & 15)) * IN_RAW;
    auto loadA = [&](int kc) {
      v16h a; const int cb = kc * 32 + ((lane & 16) ? 8 : 0);
      loadA8(rowp, cb,      IN_RAW, a, 0);
      loadA8(rowp, cb + 16, IN_RAW, a, 8);
      return a;
    };
    mlp2_core<K1, true, HD>(lane, loadA, lWs1, lb1, lWs2, lb2, lg, lbt,
                            lHs[w], lY[w], lR[w]);
    for (int i = lane; i < 16 * HD; i += 32) {
      const int row = i >> 6, col = i & 63;
      out[(size_t)(base + row) * HD + col] = lY[w][i];
    }
  }
}

// Edge MLP (192->64->64 + LN): gathers send/rec reps straight into A fragments,
// atomically scatters edge_diff into aggr (edge output itself is discarded).
__global__ void __launch_bounds__(BLOCK) edge_gnn_kernel(
    const float* __restrict__ eemb, const float* __restrict__ send_rep,
    const float* __restrict__ rec_rep,
    const int* __restrict__ sidx, const int* __restrict__ ridx,
    const float* __restrict__ W1, const float* __restrict__ b1,
    const float* __restrict__ W2, const float* __restrict__ b2,
    const float* __restrict__ g,  const float* __restrict__ bt,
    float* __restrict__ aggr)
{
  constexpr int K1 = 192;
  __shared__ __align__(32) _Float16 lWs1[K1 * HD];
  __shared__ __align__(32) _Float16 lWs2[HD * HD];
  __shared__ __align__(32) _Float16 lHs[WAVES][2 * 32 * 16];
  __shared__ float lb1[HD], lb2[HD], lg[HD], lbt[HD];
  __shared__ float lY[WAVES][16 * 64];
  __shared__ float lR[WAVES][64];
  __shared__ int   lRi[WAVES][16];

  const int tid = threadIdx.x, lane = tid & 31, w = tid >> 5;
  stage_weights<K1, HD, true>(tid, W1, b1, W2, b2, g, bt, lWs1, lb1, lWs2, lb2, lg, lbt);
  __syncthreads();

  for (int t = 0; t < TILES; ++t) {
    const int base = ((blockIdx.x * WAVES + w) * TILES + t) * 16;
    const int e  = base + (lane & 15);
    const int si = sidx[e], ri = ridx[e];
    if (lane < 16) lRi[w][lane] = ri;
    const float* pe = eemb     + (size_t)e  * HD;
    const float* ps = send_rep + (size_t)si * HD;   // L2-resident gather
    const float* pr = rec_rep  + (size_t)ri * HD;
    auto segp = [&](int c) -> const float* {
      return (c < 64) ? (pe + c) : (c < 128) ? (ps + (c - 64)) : (pr + (c - 128));
    };
    auto loadA = [&](int kc) {
      v16h a; const int cb = kc * 32 + ((lane & 16) ? 8 : 0);
      loadA8f(segp(cb),      a, 0);
      loadA8f(segp(cb + 16), a, 8);
      return a;
    };
    mlp2_core<K1, true, HD>(lane, loadA, lWs1, lb1, lWs2, lb2, lg, lbt,
                            lHs[w], lY[w], lR[w]);
    for (int i = lane; i < 16 * HD; i += 32) {
      const int row = i >> 6, col = i & 63;
      atomicAdd(&aggr[(size_t)lRi[w][row] * HD + col], lY[w][i]);
    }
  }
}

// Node update: out = rec + MLP_LN([aggr ; rec]); safe in-place (out may == rec).
__global__ void __launch_bounds__(BLOCK) node_update_kernel(
    const float* __restrict__ aggr, const float* rec,
    const float* __restrict__ W1, const float* __restrict__ b1,
    const float* __restrict__ W2, const float* __restrict__ b2,
    const float* __restrict__ g,  const float* __restrict__ bt,
    float* out)
{
  constexpr int K1 = 128;
  __shared__ __align__(32) _Float16 lWs1[K1 * HD];
  __shared__ __align__(32) _Float16 lWs2[HD * HD];
  __shared__ __align__(32) _Float16 lHs[WAVES][2 * 32 * 16];
  __shared__ float lb1[HD], lb2[HD], lg[HD], lbt[HD];
  __shared__ float lY[WAVES][16 * 64];
  __shared__ float lR[WAVES][64];

  const int tid = threadIdx.x, lane = tid & 31, w = tid >> 5;
  stage_weights<K1, HD, true>(tid, W1, b1, W2, b2, g, bt, lWs1, lb1, lWs2, lb2, lg, lbt);
  __syncthreads();

  for (int t = 0; t < TILES; ++t) {
    const int base = ((blockIdx.x * WAVES + w) * TILES + t) * 16;
    const float* pa = aggr + (size_t)(base + (lane & 15)) * HD;
    const float* pc = rec  + (size_t)(base + (lane & 15)) * HD;
    auto loadA = [&](int kc) {
      v16h a; const int cb = kc * 32 + ((lane & 16) ? 8 : 0);
      loadA8f((cb < 64) ? (pa + cb) : (pc + cb - 64),           a, 0);
      loadA8f((cb + 16 < 64) ? (pa + cb + 16) : (pc + cb - 48), a, 8);
      return a;
    };
    mlp2_core<K1, true, HD>(lane, loadA, lWs1, lb1, lWs2, lb2, lg, lbt,
                            lHs[w], lY[w], lR[w]);
    for (int i = lane; i < 16 * HD; i += 32) {
      const int row = i >> 6, col = i & 63;
      const size_t idx = (size_t)(base + row) * HD + col;
      out[idx] = rec[idx] + lY[w][i];   // read-before-write per element
    }
  }
}

// Output map: 64 -> 64 (SiLU) -> 17, no LN. Layer-2 padded to 32 cols.
__global__ void __launch_bounds__(BLOCK) output_mlp_kernel(
    const float* __restrict__ x,
    const float* __restrict__ W1, const float* __restrict__ b1,
    const float* __restrict__ W2, const float* __restrict__ b2,
    float* __restrict__ out)
{
  constexpr int K1 = 64, NOUT = 17;
  __shared__ __align__(32) _Float16 lWs1[K1 * HD];
  __shared__ __align__(32) _Float16 lWs2[HD * 32];
  __shared__ __align__(32) _Float16 lHs[WAVES][2 * 32 * 16];
  __shared__ float lb1[HD], lb2[HD];
  __shared__ float lY[WAVES][16 * 64];
  __shared__ float lR[WAVES][64];

  const int tid = threadIdx.x, lane = tid & 31, w = tid >> 5;
  stage_weights<K1, NOUT, false>(tid, W1, b1, W2, b2, nullptr, nullptr,
                                 lWs1, lb1, lWs2, lb2, nullptr, nullptr);
  __syncthreads();

  for (int t = 0; t < TILES; ++t) {
    const int base = ((blockIdx.x * WAVES + w) * TILES + t) * 16;
    const float* rowp = x + (size_t)(base + (lane & 15)) * HD;
    auto loadA = [&](int kc) {
      v16h a; const int cb = kc * 32 + ((lane & 16) ? 8 : 0);
      loadA8f(rowp + cb,      a, 0);
      loadA8f(rowp + cb + 16, a, 8);
      return a;
    };
    mlp2_core<K1, false, NOUT>(lane, loadA, lWs1, lb1, lWs2, lb2, nullptr, nullptr,
                               lHs[w], lY[w], lR[w]);
    for (int i = lane; i < 16 * 32; i += 32) {
      const int row = i >> 5, col = i & 31;
      if (col < NOUT) out[(size_t)(base + row) * NOUT + col] = lY[w][row * 64 + col];
    }
  }
}

// ============================ host driver ===================================

extern "C" void kernel_launch(void* const* d_in, const int* in_sizes, int n_in,
                              void* d_out, int out_size, void* d_ws, size_t ws_size,
                              hipStream_t stream)
{
  (void)in_sizes; (void)n_in; (void)out_size; (void)ws_size;
  constexpr int NG = 131072, NM = 32768, E = 262144;
  constexpr int RPB = WAVES * TILES * 16;   // rows per block = 256

  const float* grid_x = (const float*)d_in[0];
  const float* g2m_x  = (const float*)d_in[1];
  const float* m2g_x  = (const float*)d_in[2];
  const float* m2m_x  = (const float*)d_in[3];
  const float* mesh_x = (const float*)d_in[4];
  const int*   g2m_ei = (const int*)d_in[5];   // [0]=send (grid), [1]=rec (mesh)
  const int*   m2g_ei = (const int*)d_in[6];   // [0]=send (mesh), [1]=rec (grid)
  const int*   m2m_ei = (const int*)d_in[7];

  auto P = [&](int i) { return (const float*)d_in[8 + i]; };
  struct Mlp { const float *W1, *b1, *W2, *b2, *g, *bt; };
  auto mlp6 = [&](int o) { return Mlp{P(o), P(o+1), P(o+2), P(o+3), P(o+4), P(o+5)}; };
  const Mlp grid_emb_p = mlp6(0);
  const Mlp mesh_p     = mlp6(6);
  const Mlp g2m_p      = mlp6(12);
  const Mlp m2g_p      = mlp6(18);
  const Mlp m2m_p      = mlp6(24);
  const Mlp enc_p      = mlp6(30);
  const Mlp outmap_p   = {P(36), P(37), P(38), P(39), nullptr, nullptr};
  const Mlp g2m_edge   = mlp6(40), g2m_aggr = mlp6(46);
  const Mlp m2g_edge   = mlp6(52), m2g_aggr = mlp6(58);
  const Mlp prc_edge   = mlp6(64), prc_aggr = mlp6(70);

  // Workspace (~176 MB); node reps stay L2-resident (192MB L2).
  float* GE = (float*)d_ws;
  float* GR = GE + (size_t)NG * 64;
  float* AG = GR + (size_t)NG * 64;
  float* E1 = AG + (size_t)NG * 64;
  float* M1 = E1 + (size_t)E  * 64;

  const dim3 blk(BLOCK);
#define MLPA(m) (m).W1, (m).b1, (m).W2, (m).b2, (m).g, (m).bt

  // --- embed ---
  node_mlp_kernel<18, 32><<<NG / RPB, blk, 0, stream>>>(grid_x, MLPA(grid_emb_p), GE);
  node_mlp_kernel< 3, 32><<<E  / RPB, blk, 0, stream>>>(g2m_x,  MLPA(g2m_p),      E1);
  node_mlp_kernel< 3, 32><<<NM / RPB, blk, 0, stream>>>(mesh_x, MLPA(mesh_p),     M1);

  // --- encode: grid -> mesh ---
  hipMemsetAsync(AG, 0, (size_t)NM * 64 * sizeof(float), stream);
  edge_gnn_kernel<<<E / RPB, blk, 0, stream>>>(E1, GE, M1, g2m_ei, g2m_ei + E,
                                               MLPA(g2m_edge), AG);
  node_update_kernel<<<NM / RPB, blk, 0, stream>>>(AG, M1, MLPA(g2m_aggr), M1);
  node_mlp_kernel<64, 64><<<NG / RPB, blk, 0, stream>>>(GE, MLPA(enc_p), GR);

  // --- process on mesh (processor edge outputs are discarded by caller) ---
  node_mlp_kernel<4, 32><<<E / RPB, blk, 0, stream>>>(m2m_x, MLPA(m2m_p), E1);
  hipMemsetAsync(AG, 0, (size_t)NM * 64 * sizeof(float), stream);
  edge_gnn_kernel<<<E / RPB, blk, 0, stream>>>(E1, M1, M1, m2m_ei, m2m_ei + E,
                                               MLPA(prc_edge), AG);
  node_update_kernel<<<NM / RPB, blk, 0, stream>>>(AG, M1, MLPA(prc_aggr), M1);

  // --- decode: mesh -> grid ---
  node_mlp_kernel<3, 32><<<E / RPB, blk, 0, stream>>>(m2g_x, MLPA(m2g_p), E1);
  hipMemsetAsync(AG, 0, (size_t)NG * 64 * sizeof(float), stream);
  edge_gnn_kernel<<<E / RPB, blk, 0, stream>>>(E1, M1, GR, m2g_ei, m2g_ei + E,
                                               MLPA(m2g_edge), AG);
  node_update_kernel<<<NG / RPB, blk, 0, stream>>>(AG, GR, MLPA(m2g_aggr), GR);

  // --- output map ---
  output_mlp_kernel<<<NG / RPB, blk, 0, stream>>>(GR, outmap_p.W1, outmap_p.b1,
                                                  outmap_p.W2, outmap_p.b2,
                                                  (float*)d_out);
#undef MLPA
}
